// MultiHeadAttention_55448027791771
// MI455X (gfx1250) — matrix operand order repaired and checked by reading, not compiled, not run on hardware
//
#include <hip/hip_runtime.h>
#include <math.h>

typedef _Float16 half_t;
typedef _Float16 v8h  __attribute__((ext_vector_type(8)));
typedef _Float16 v16h __attribute__((ext_vector_type(16)));
typedef float    v8f  __attribute__((ext_vector_type(8)));

#define WMMA_F32_F16(A, B, C) \
  __builtin_amdgcn_wmma_f32_16x16x32_f16(false, (A), false, (B), (short)0, (C), false, false)

static __device__ __forceinline__ v16h cat8(v8h lo, v8h hi) {
  return __builtin_shufflevector(lo, hi, 0,1,2,3,4,5,6,7,8,9,10,11,12,13,14,15);
}

constexpr int SEQ = 2048;
constexpr int NH  = 16;
constexpr int HD  = 64;
constexpr int DM  = 1024;

// ---------------------------------------------------------------------------
// elementwise f32 -> f16
__global__ void k_cvt_f16(const float* __restrict__ in, half_t* __restrict__ out, int n) {
  int i = blockIdx.x * blockDim.x + threadIdx.x;
  if (i < n) out[i] = (half_t)in[i];
}

// W[k][n] (f32, row-major KxN) -> Wt[n][k] (f16, row-major NxK)
__global__ void k_transpose_w(const float* __restrict__ W, half_t* __restrict__ Wt) {
  int i = blockIdx.x * blockDim.x + threadIdx.x;    // output-linear [n][k]
  int n = i >> 10;
  int k = i & (DM - 1);
  Wt[i] = (half_t)W[(size_t)k * DM + n];
}

// ---------------------------------------------------------------------------
// C[m][n] = sum_k A[m][k]*Bt[n][k] + bias[n]
// A: f16 row-major MxK ; Bt: f16 row-major NxK (pre-transposed weights)
// one wave -> 16(M) x 64(N) output strip (4 accumulators, A-frag reused 4x);
// 8 waves/block tile N -> block covers 16 x 512.
template <typename OutT>
__global__ __launch_bounds__(256) void k_gemm16(const half_t* __restrict__ A,
                                                const half_t* __restrict__ Bt,
                                                const float* __restrict__ bias,
                                                OutT* __restrict__ C,
                                                int M, int N, int K) {
  const int lane = threadIdx.x & 31;
  const int w    = threadIdx.x >> 5;
  const int ln   = lane & 15;
  const int g    = lane >> 4;
  const int n0   = (blockIdx.x * 8 + w) * 64;
  const int m0   = blockIdx.y * 16;

  const half_t* arow = A + (size_t)(m0 + ln) * K;        // A-frag: lane -> M row
  const half_t* br0 = Bt + (size_t)(n0 + ln) * K;        // B-frag t: lane -> row n0+16t+ln
  const half_t* br1 = br0 + (size_t)16 * K;
  const half_t* br2 = br1 + (size_t)16 * K;
  const half_t* br3 = br2 + (size_t)16 * K;

  v8f acc0 = {}, acc1 = {}, acc2 = {}, acc3 = {};
#pragma unroll 2
  for (int k0 = 0; k0 < K; k0 += 32) {
    // A 16x32 f16 layout: half i -> K = k0 + 16g + i (contiguous, matches B)
    v16h a = cat8(*(const v8h*)(arow + k0 + 16 * g),
                  *(const v8h*)(arow + k0 + 16 * g + 8));
    // B 32x16 f16 layout: half i -> K = k0 + 16g + i (contiguous 32B per lane)
    v16h b0 = *(const v16h*)(br0 + k0 + 16 * g);
    v16h b1 = *(const v16h*)(br1 + k0 + 16 * g);
    v16h b2 = *(const v16h*)(br2 + k0 + 16 * g);
    v16h b3 = *(const v16h*)(br3 + k0 + 16 * g);
    acc0 = WMMA_F32_F16(a, b0, acc0);
    acc1 = WMMA_F32_F16(a, b1, acc1);
    acc2 = WMMA_F32_F16(a, b2, acc2);
    acc3 = WMMA_F32_F16(a, b3, acc3);
  }
  const float bn0 = bias[n0 + ln];
  const float bn1 = bias[n0 + 16 + ln];
  const float bn2 = bias[n0 + 32 + ln];
  const float bn3 = bias[n0 + 48 + ln];
#pragma unroll
  for (int r = 0; r < 8; ++r) {                          // C: vgpr r -> row r+8g, lane -> col
    OutT* crow = C + (size_t)(m0 + r + 8 * g) * N + n0 + ln;
    crow[0]  = (OutT)(acc0[r] + bn0);
    crow[16] = (OutT)(acc1[r] + bn1);
    crow[32] = (OutT)(acc2[r] + bn2);
    crow[48] = (OutT)(acc3[r] + bn3);
  }
}

// ---------------------------------------------------------------------------
// RoPE on full 1024-dim vectors (matches reference exactly, incl. the
// even*sin - odd*cos second half), scatter into per-head [b][h][pos][64] f16.
__global__ void k_rope(const half_t* __restrict__ Qp, const half_t* __restrict__ Kp,
                       half_t* __restrict__ Qh, half_t* __restrict__ Kh) {
  int i = blockIdx.x * blockDim.x + threadIdx.x;     // [b][pos][j<512]
  int j   = i & 511;
  int pos = (i >> 9) & (SEQ - 1);
  int b   = i >> 20;
  // freq = 10000^(-2j/1024) = exp2(-j*log2(10000)/512)
  float f   = exp2f((float)j * (-13.287712379549449f / 512.0f));
  float ang = (float)pos * f;
  float s, c;
  sincosf(ang, &s, &c);
  size_t base = (size_t)(b * SEQ + pos) * DM;
  int d0 = j, d1 = 512 + j;
  size_t o0q = ((size_t)(b * NH + (d0 >> 6)) * SEQ + pos) * HD + (d0 & 63);
  size_t o1q = ((size_t)(b * NH + (d1 >> 6)) * SEQ + pos) * HD + (d1 & 63);
  {
    float ev = (float)Qp[base + 2 * j], od = (float)Qp[base + 2 * j + 1];
    Qh[o0q] = (half_t)(ev * c - od * s);
    Qh[o1q] = (half_t)(ev * s - od * c);
  }
  {
    float ev = (float)Kp[base + 2 * j], od = (float)Kp[base + 2 * j + 1];
    Kh[o0q] = (half_t)(ev * c - od * s);
    Kh[o1q] = (half_t)(ev * s - od * c);
  }
}

// V[b][pos][1024] f16 -> Vt[b][h][64][2048] f16 (transposed per head)
__global__ void k_vfmt(const half_t* __restrict__ Vp, half_t* __restrict__ Vt) {
  int i = blockIdx.x * blockDim.x + threadIdx.x;     // [b][pos][d]
  int d   = i & (DM - 1);
  int pos = (i >> 10) & (SEQ - 1);
  int b   = i >> 21;
  int bh  = b * NH + (d >> 6);
  Vt[((size_t)bh * HD + (d & 63)) * SEQ + pos] = Vp[i];
}

// ---------------------------------------------------------------------------
// Flash attention: one wave = 16 queries of one (b,h); online softmax;
// ctx written as f16 rows [b][q][h*64+d] ready to be the A matrix of the
// output projection.
__global__ __launch_bounds__(256) void k_flash(const half_t* __restrict__ Qh,
                                               const half_t* __restrict__ Kh,
                                               const half_t* __restrict__ Vt,
                                               half_t* __restrict__ ctx) {
  __shared__ __align__(128) half_t pbuf[8][16][40];  // 80B row stride: 16B-aligned, bank-spread
  const int lane = threadIdx.x & 31;
  const int w    = threadIdx.x >> 5;
  const int ln   = lane & 15;
  const int g    = lane >> 4;
  const int bh   = blockIdx.y;
  const int b    = bh >> 4;
  const int h    = bh & 15;
  const int q0   = (blockIdx.x * 8 + w) * 16;

  const half_t* qrow  = Qh + ((size_t)bh * SEQ + q0 + ln) * HD;
  const half_t* kbase = Kh + (size_t)bh * SEQ * HD;
  const half_t* vbase = Vt + (size_t)bh * HD * SEQ;

  // Q A-fragments for kk=0..31 and kk=32..63 (reused across all key tiles)
  v16h aq0 = cat8(*(const v8h*)(qrow + 16 * g),      *(const v8h*)(qrow + 16 * g + 8));
  v16h aq1 = cat8(*(const v8h*)(qrow + 32 + 16 * g), *(const v8h*)(qrow + 40 + 16 * g));

  v8f acc0 = {}, acc1 = {}, acc2 = {}, acc3 = {};
  float mrun[8], lrun[8];
#pragma unroll
  for (int r = 0; r < 8; ++r) { mrun[r] = -INFINITY; lrun[r] = 0.0f; }

  for (int k0 = 0; k0 < q0 + 16; k0 += 32) {
    // ---- scores: two 16x16 tiles (keys k0..k0+15 and k0+16..k0+31)
    const half_t* kr0 = kbase + (size_t)(k0 + ln) * HD;
    const half_t* kr1 = kbase + (size_t)(k0 + 16 + ln) * HD;
    v8f s0 = {}, s1 = {};
    s0 = WMMA_F32_F16(aq0, *(const v16h*)(kr0 + 16 * g),      s0);
    s0 = WMMA_F32_F16(aq1, *(const v16h*)(kr0 + 32 + 16 * g), s0);
    s1 = WMMA_F32_F16(aq0, *(const v16h*)(kr1 + 16 * g),      s1);
    s1 = WMMA_F32_F16(aq1, *(const v16h*)(kr1 + 32 + 16 * g), s1);

    // ---- scale + causal mask + online softmax (rows live at vgpr r, half-wave g)
#pragma unroll
    for (int r = 0; r < 8; ++r) {
      int row = q0 + r + 8 * g;
      float a = ((k0 + ln)      <= row) ? s0[r] * 0.125f : -INFINITY;
      float c = ((k0 + 16 + ln) <= row) ? s1[r] * 0.125f : -INFINITY;
      float v = fmaxf(a, c);
      v = fmaxf(v, __shfl_xor(v, 1));
      v = fmaxf(v, __shfl_xor(v, 2));
      v = fmaxf(v, __shfl_xor(v, 4));
      v = fmaxf(v, __shfl_xor(v, 8));
      float m2 = fmaxf(mrun[r], v);        // finite after first tile (col 0 always valid)
      float e0 = __expf(a - m2);
      float e1 = __expf(c - m2);
      float rs = e0 + e1;
      rs += __shfl_xor(rs, 1);
      rs += __shfl_xor(rs, 2);
      rs += __shfl_xor(rs, 4);
      rs += __shfl_xor(rs, 8);
      float cf = __expf(mrun[r] - m2);
      lrun[r] = lrun[r] * cf + rs;
      mrun[r] = m2;
      acc0[r] *= cf; acc1[r] *= cf; acc2[r] *= cf; acc3[r] *= cf;
      // stage P (C-layout) into LDS row-major [16][32]
      pbuf[w][r + 8 * g][ln]      = (half_t)e0;
      pbuf[w][r + 8 * g][16 + ln] = (half_t)e1;
    }
    asm volatile("s_wait_dscnt 0x0" ::: "memory");

    // ---- P as A-fragment (16x32) from LDS
    v16h ap = cat8(*(const v8h*)&pbuf[w][ln][16 * g],
                   *(const v8h*)&pbuf[w][ln][16 * g + 8]);

    // ---- P @ V : Vt rows are d, halves are consecutive k_seq (contiguous 32B)
    const half_t* vr = vbase + k0 + 16 * g;
    acc0 = WMMA_F32_F16(ap, *(const v16h*)(vr + (size_t)(ln)      * SEQ), acc0);
    acc1 = WMMA_F32_F16(ap, *(const v16h*)(vr + (size_t)(16 + ln) * SEQ), acc1);
    acc2 = WMMA_F32_F16(ap, *(const v16h*)(vr + (size_t)(32 + ln) * SEQ), acc2);
    acc3 = WMMA_F32_F16(ap, *(const v16h*)(vr + (size_t)(48 + ln) * SEQ), acc3);
  }

  // ---- normalize + store ctx as f16 rows [b][q][h*64 + d]
  half_t* cb = ctx + (size_t)b * SEQ * DM + (size_t)h * HD + ln;
#pragma unroll
  for (int r = 0; r < 8; ++r) {
    int row = q0 + r + 8 * g;
    float inv = 1.0f / lrun[r];
    half_t* crow = cb + (size_t)row * DM;
    crow[0]  = (half_t)(acc0[r] * inv);
    crow[16] = (half_t)(acc1[r] * inv);
    crow[32] = (half_t)(acc2[r] * inv);
    crow[48] = (half_t)(acc3[r] * inv);
  }
}

// ---------------------------------------------------------------------------
extern "C" void kernel_launch(void* const* d_in, const int* in_sizes, int n_in,
                              void* d_out, int out_size, void* d_ws, size_t ws_size,
                              hipStream_t stream) {
  const float* x  = (const float*)d_in[0];
  const float* Wq = (const float*)d_in[1];
  const float* bq = (const float*)d_in[2];
  const float* Wk = (const float*)d_in[3];
  const float* bk = (const float*)d_in[4];
  const float* Wv = (const float*)d_in[5];
  const float* bv = (const float*)d_in[6];
  const float* Wo = (const float*)d_in[7];
  const float* bo = (const float*)d_in[8];
  float* out = (float*)d_out;
  char* ws = (char*)d_ws;

  constexpr size_t MROWS = (size_t)2 * SEQ;          // 4096
  constexpr size_t SZ_MAT16 = MROWS * DM * 2;        // 8 MB  (f16 4096x1024)
  constexpr size_t SZ_W16   = (size_t)DM * DM * 2;   // 2 MB

  half_t* Xh  = (half_t*)(ws);
  half_t* Wtq = (half_t*)(ws + SZ_MAT16);
  half_t* Wtk = Wtq + (size_t)DM * DM;
  half_t* Wtv = Wtk + (size_t)DM * DM;
  half_t* Wto = Wtv + (size_t)DM * DM;
  half_t* Qp  = (half_t*)(ws + SZ_MAT16 + 4 * SZ_W16);
  half_t* Kp  = Qp + MROWS * DM;
  half_t* Vp  = Kp + MROWS * DM;
  half_t* Qh  = Vp + MROWS * DM;                     // [b][h][seq][64]
  half_t* Kh  = Qh + MROWS * DM;                     // [b][h][seq][64]
  half_t* Vt  = Kh + MROWS * DM;                     // [b][h][64][seq]
  half_t* Ch  = Vt + MROWS * DM;                     // [b][q][1024]
  // total workspace: 8 + 8 + 7*8 = 72 MB

  const int nX = (int)(MROWS * DM);                  // 4194304
  k_cvt_f16<<<nX / 256, 256, 0, stream>>>(x, Xh, nX);
  k_transpose_w<<<(DM * DM) / 256, 256, 0, stream>>>(Wq, Wtq);
  k_transpose_w<<<(DM * DM) / 256, 256, 0, stream>>>(Wk, Wtk);
  k_transpose_w<<<(DM * DM) / 256, 256, 0, stream>>>(Wv, Wtv);
  k_transpose_w<<<(DM * DM) / 256, 256, 0, stream>>>(Wo, Wto);

  dim3 gg(DM / 512, MROWS / 16);                     // (2, 256)
  k_gemm16<half_t><<<gg, 256, 0, stream>>>(Xh, Wtq, bq, Qp, (int)MROWS, DM, DM);
  k_gemm16<half_t><<<gg, 256, 0, stream>>>(Xh, Wtk, bk, Kp, (int)MROWS, DM, DM);
  k_gemm16<half_t><<<gg, 256, 0, stream>>>(Xh, Wtv, bv, Vp, (int)MROWS, DM, DM);

  k_rope<<<(2 * SEQ * 512) / 256, 256, 0, stream>>>(Qp, Kp, Qh, Kh);
  k_vfmt<<<nX / 256, 256, 0, stream>>>(Vp, Vt);

  k_flash<<<dim3(SEQ / 128, 2 * NH), 256, 0, stream>>>(Qh, Kh, Vt, Ch);

  k_gemm16<float><<<gg, 256, 0, stream>>>(Ch, Wto, bo, out, (int)MROWS, DM, DM);
}
